// HiddenElementsFinalLayer_fullyconnected_81106162418204
// MI455X (gfx1250) — matrix operand order repaired and checked by reading, not compiled, not run on hardware
//
#include <hip/hip_runtime.h>

// Problem constants (match the reference)
#define KTOT   512                    // HIDDEN_SIZE
#define NTOT   (512 * 64)             // HIDDEN_SIZE * MAP_ELEMENT_SIZE = 32768
#define MTOT   1024                   // BATCH

// Tiling
#define BM   128
#define BN   128
#define BK   64
#define NK   (KTOT / BK)              // 8 K-slices
#define LSTR 72                       // bf16 elems per LDS row: 64 + 8 pad (144 B, 16B-aligned)

typedef __attribute__((ext_vector_type(16))) __bf16 v16bf;
typedef __attribute__((ext_vector_type(8)))  __bf16 v8bf;
typedef __attribute__((ext_vector_type(8)))  float  v8f;
typedef __attribute__((ext_vector_type(4)))  float  fx4;

__global__ __launch_bounds__(256, 1)
void fused_linear_bias_wmma(const float* __restrict__ X,
                            const float* __restrict__ W,
                            const float* __restrict__ Bias,
                            float* __restrict__ Out) {
    // Double-buffered bf16 staging tiles: 2 * (128*72 + 128*72) * 2B = 72 KB
    __shared__ __align__(16) __bf16 As[2][BM * LSTR];
    __shared__ __align__(16) __bf16 Bs[2][BN * LSTR];

    const int tid    = threadIdx.x;
    const int lane   = tid & 31;
    const int wave   = tid >> 5;          // 0..7
    const int wave_m = wave >> 2;         // 0..1 : 64-row band
    const int wave_n = wave & 3;          // 0..3 : 32-col band

    const int block_m = blockIdx.y * BM;
    const int block_n = blockIdx.x * BN;

    // Staging: thread t loads row (t>>1), 32 consecutive fp32 at col (t&1)*32
    const int ldr = tid >> 1;             // 0..127
    const int ldc = (tid & 1) * 32;       // 0 or 32

    const int m_lane = lane & 15;         // M / N index within a 16x16 tile
    const int khalf  = lane >> 4;         // 0 or 1 : which K half this lane holds

    const float* gA = X + (size_t)(block_m + ldr) * KTOT + ldc;
    const float* gB = W + (size_t)(block_n + ldr) * KTOT + ldc;

    v8f acc[4][2] = {};
    fx4 rA[8], rB[8];

    // ---- Prologue: load + stage K-slice 0 into buffer 0 ----
    #pragma unroll
    for (int i = 0; i < 8; ++i) {
        rA[i] = ((const fx4*)gA)[i];
        rB[i] = ((const fx4*)gB)[i];
    }
    {
        __bf16* sa = &As[0][ldr * LSTR + ldc];
        __bf16* sb = &Bs[0][ldr * LSTR + ldc];
        #pragma unroll
        for (int i = 0; i < 8; i += 2) {
            v8bf pa, pb;
            #pragma unroll
            for (int j = 0; j < 4; ++j) {
                pa[j]     = (__bf16)rA[i][j];
                pa[j + 4] = (__bf16)rA[i + 1][j];
                pb[j]     = (__bf16)rB[i][j];
                pb[j + 4] = (__bf16)rB[i + 1][j];
            }
            *(v8bf*)(sa + i * 4) = pa;
            *(v8bf*)(sb + i * 4) = pb;
        }
    }

    // ---- Main pipeline: one barrier per K-slice, double-buffered LDS ----
    for (int kt = 0; kt < NK; ++kt) {
        const int buf = kt & 1;
        __syncthreads();

        // Issue next slice's global loads early (latency hidden under WMMAs)
        if (kt + 1 < NK) {
            const float* ga = gA + (kt + 1) * BK;
            const float* gb = gB + (kt + 1) * BK;
            #pragma unroll
            for (int i = 0; i < 8; ++i) {
                rA[i] = ((const fx4*)ga)[i];
                rB[i] = ((const fx4*)gb)[i];
            }
            if (kt + 2 < NK) {            // emits global_prefetch_b8
                __builtin_prefetch(gA + (kt + 2) * BK, 0, 3);
                __builtin_prefetch(gB + (kt + 2) * BK, 0, 3);
            }
        }

        // Compute on LDS[buf]: 2 WMMA K-steps of 32
        #pragma unroll
        for (int ks = 0; ks < 2; ++ks) {
            const int koff = ks * 32;

            // A fragments (16x32 bf16, ISA layout):
            // lane m = m_lane; e0..7 = K[koff+khalf*8 ..], e8..15 = K[koff+16+khalf*8 ..]
            v16bf afr[4];
            #pragma unroll
            for (int wm = 0; wm < 4; ++wm) {
                const int m = wave_m * 64 + wm * 16 + m_lane;
                const __bf16* p = &As[buf][m * LSTR + koff + khalf * 8];
                union { v8bf u[2]; v16bf v; } cv;
                cv.u[0] = *(const v8bf*)p;
                cv.u[1] = *(const v8bf*)(p + 16);
                afr[wm] = cv.v;
            }

            // B fragments (32x16 bf16): lane n = m_lane; e0..7 = K[koff+khalf*8 ..], e8..15 = K[koff+16+khalf*8 ..]
            v16bf bfr[2];
            #pragma unroll
            for (int wn = 0; wn < 2; ++wn) {
                const int n = wave_n * 32 + wn * 16 + m_lane;
                const __bf16* p = &Bs[buf][n * LSTR + koff + khalf * 8];
                union { v8bf u[2]; v16bf v; } cv;
                cv.u[0] = *(const v8bf*)p;
                cv.u[1] = *(const v8bf*)(p + 16);
                bfr[wn] = cv.v;
            }

            #pragma unroll
            for (int wm = 0; wm < 4; ++wm) {
                #pragma unroll
                for (int wn = 0; wn < 2; ++wn) {
                    acc[wm][wn] = __builtin_amdgcn_wmma_f32_16x16x32_bf16(
                        /*neg_a=*/false, afr[wm],
                        /*neg_b=*/false, bfr[wn],
                        /*c_mod=*/(short)0, acc[wm][wn],
                        /*reuse_a=*/false, /*reuse_b=*/false);
                }
            }
        }

        // Stage next slice into the other buffer (safe: that buffer's readers
        // finished before the barrier at the top of this iteration)
        if (kt + 1 < NK) {
            __bf16* sa = &As[buf ^ 1][ldr * LSTR + ldc];
            __bf16* sb = &Bs[buf ^ 1][ldr * LSTR + ldc];
            #pragma unroll
            for (int i = 0; i < 8; i += 2) {
                v8bf pa, pb;
                #pragma unroll
                for (int j = 0; j < 4; ++j) {
                    pa[j]     = (__bf16)rA[i][j];
                    pa[j + 4] = (__bf16)rA[i + 1][j];
                    pb[j]     = (__bf16)rB[i][j];
                    pb[j + 4] = (__bf16)rB[i + 1][j];
                }
                *(v8bf*)(sa + i * 4) = pa;
                *(v8bf*)(sb + i * 4) = pb;
            }
        }
    }

    // ---- Epilogue: C/D layout VGPR r -> M = r (+8 for lane>=16), N = lane&15
    #pragma unroll
    for (int wn = 0; wn < 2; ++wn) {
        const int n = block_n + wave_n * 32 + wn * 16 + m_lane;
        const float bias = Bias[n];
        #pragma unroll
        for (int wm = 0; wm < 4; ++wm) {
            const int mbase = block_m + wave_m * 64 + wm * 16 + khalf * 8;
            #pragma unroll
            for (int r = 0; r < 8; ++r) {
                Out[(size_t)(mbase + r) * NTOT + n] = acc[wm][wn][r] + bias;
            }
        }
    }
}

extern "C" void kernel_launch(void* const* d_in, const int* in_sizes, int n_in,
                              void* d_out, int out_size, void* d_ws, size_t ws_size,
                              hipStream_t stream) {
    (void)in_sizes; (void)n_in; (void)out_size; (void)d_ws; (void)ws_size;
    const float* X    = (const float*)d_in[0];   // [1024, 512]
    const float* W    = (const float*)d_in[1];   // [32768, 512]
    const float* Bias = (const float*)d_in[2];   // [32768]
    float* Out        = (float*)d_out;           // [1024, 32768]

    dim3 grid(NTOT / BN, MTOT / BM);             // (256, 8)
    dim3 block(256);
    fused_linear_bias_wmma<<<grid, block, 0, stream>>>(X, W, Bias, Out);
}
